// MyModel_41755672051997
// MI455X (gfx1250) — matrix operand order repaired and checked by reading, not compiled, not run on hardware
//
#include <hip/hip_runtime.h>
#include <hip/hip_bf16.h>
#include <math.h>

// ---------------------------------------------------------------------------
// Problem constants
// ---------------------------------------------------------------------------
#define HMAP    160
#define WMAP    160
#define MROWS   (HMAP * WMAP)      // 25600
#define KDIM    1024
#define CH      196                // channels per map (4*7*7)
#define NCOLS   392                // bb_map cols [0,196) ++ ps_map cols [196,392)
#define NPAD    400                // padded to 25 tiles of 16
#define NROIS   300
#define NALL    600
#define KK      7
#define HH      14                 // K*S
#define NSAMP   196                // 14*14 sample points per ROI
#define IMM1    1279.0f            // IM - 1
#define IOU_T   0.3f
#define MAXOUT  300

// staging geometry (per 256-thread block, per 32-K chunk)
#define WCHUNK  6272               // 32*196 floats: one matrix's contiguous chunk
#define BSTG    49                 // 2*WCHUNK/256
#define ASTG    4                  // 32*32/256

// d_out layout (flat, reference return order)
#define OFF_CLS    0               // [600,2]
#define OFF_CLSR   1200            // [600]
#define OFF_SCORE  1800            // [600]
#define OFF_MASK   2400            // [600,14,14,2]
#define OFF_ROIS   237600          // [600,5]
#define OFF_BBOX   240600          // [600,4]
#define OFF_KEEP   243000          // [300]

// ---------------------------------------------------------------------------
// Vector types for WMMA
// ---------------------------------------------------------------------------
typedef __attribute__((ext_vector_type(16))) __bf16         bf16x16;
typedef __attribute__((ext_vector_type(8)))  float          floatx8;
typedef __attribute__((ext_vector_type(8)))  unsigned short u16x8;

__device__ __forceinline__ unsigned short f2bf(float f) {
    union { float f; unsigned u; } x; x.f = f;
    unsigned u = x.u;
    // round-to-nearest-even truncation to bf16
    return (unsigned short)((u + 0x7FFFu + ((u >> 16) & 1u)) >> 16);
}

__device__ __forceinline__ bf16x16 frag_cat(u16x8 lo, u16x8 hi) {
    return __builtin_bit_cast(bf16x16,
        __builtin_shufflevector(lo, hi, 0,1,2,3,4,5,6,7,8,9,10,11,12,13,14,15));
}

// ---------------------------------------------------------------------------
// Kernel A: fused dual GEMM  maps[25600, 392] = feat @ [W_bb | W_ps] + bias
// bf16 WMMA (v_wmma_f32_16x16x32_bf16), f32 accumulate.
// Block = 256 thr = 8 waves; block stripe = 32 rows x 392 cols.
//   wave w: m_tile = w&1 (16 rows), n_group = w>>1 (tiles 6/6/6/7 of 16 cols)
// Software pipeline: next K-chunk is global-loaded into registers while the
// current chunk is consumed from LDS. Staged loads are AFFINE (one base +
// immediate offsets) so no per-load address registers; [n][kk] scatter math
// happens only in the post-barrier LDS-store phase. B fragments are register
// double-buffered so dscnt waits stay partial.
// ---------------------------------------------------------------------------
__global__ __launch_bounds__(256) void gemm_maps_kernel(
    const float* __restrict__ feat, const float* __restrict__ Wbb,
    const float* __restrict__ Wps,  const float* __restrict__ bbb,
    const float* __restrict__ bps,  float* __restrict__ maps) {

    __shared__ __align__(16) unsigned short Alds[32 * 40];    // 32 rows x 32 K (pad 40)
    __shared__ __align__(16) unsigned short Blds[NPAD * 40];  // 400 cols x 32 K (pad 40)

    const int tid  = threadIdx.x;
    const int m0   = blockIdx.x * 32;
    const int wave = tid >> 5;
    const int lane = tid & 31;
    const int mt   = wave & 1;                 // m tile in block
    const int ng   = wave >> 1;                // n group 0..3
    const int t0   = ng * 6;
    const int tcnt = (ng == 3) ? 7 : 6;
    const int lm   = lane & 15;                // column / row-within-tile lane
    const int half = lane >> 4;                // K-half selector per ISA layout

    const int arow = tid >> 5;                 // A-stage base row (0..7)
    const int akk  = tid & 31;                 // A-stage K within chunk

    floatx8 acc[7];
#pragma unroll
    for (int t = 0; t < 7; ++t) acc[t] = (floatx8)0.0f;

    // ---- prologue: constant zero pad (cols 392..399), then stage chunk 0 ----
    Blds[(392 + (tid >> 5)) * 40 + (tid & 31)] = 0;   // 8 rows x 32 kk, once
#pragma unroll
    for (int j = 0; j < BSTG; ++j) {
        const int gi  = tid + 256 * j;                 // flat stream over 2x6272
        const int rem = (gi < WCHUNK) ? gi : gi - WCHUNK;
        const int kk  = rem / CH;
        const int n   = rem - kk * CH + ((gi < WCHUNK) ? 0 : CH);
        const float v = (gi < WCHUNK) ? Wbb[gi] : Wps[gi - WCHUNK];
        Blds[n * 40 + kk] = f2bf(v);
    }
#pragma unroll
    for (int j = 0; j < ASTG; ++j)
        Alds[(arow + 8 * j) * 40 + akk] =
            f2bf(feat[(size_t)(m0 + arow + 8 * j) * KDIM + akk]);
    __syncthreads();

    for (int k0 = 0; k0 < KDIM; k0 += 32) {
        const bool hasNext = (k0 + 32) < KDIM;

        // ---- issue next chunk's global loads (affine: base + imm offsets) ----
        float bst[BSTG];
        float ast[ASTG];
        if (hasNext) {
            const int kn = k0 + 32;
            const float* wbase0 = Wbb + (size_t)kn * CH;   // contiguous 6272 floats
            const float* wbase1 = Wps + (size_t)kn * CH;
#pragma unroll
            for (int j = 0; j < BSTG; ++j) {
                const int gi = tid + 256 * j;
                bst[j] = (gi < WCHUNK) ? wbase0[gi] : wbase1[gi - WCHUNK];
            }
            const float* fbase = feat + (size_t)(m0 + arow) * KDIM + kn + akk;
#pragma unroll
            for (int j = 0; j < ASTG; ++j)
                ast[j] = fbase[(size_t)(8 * j) * KDIM];
        }

        // ---- compute current chunk from LDS ----
        // A fragment (ISA 16-bit A 16x32 layout)
        const unsigned short* ap = Alds + (mt * 16 + lm) * 40 + half * 8;
        u16x8 alo = *(const u16x8*)(ap);        // K = half*8 + 0..7
        u16x8 ahi = *(const u16x8*)(ap + 16);   // K = 16 + half*8 + 0..7
        const bf16x16 afrag = frag_cat(alo, ahi);

        // B fragments, register double-buffered
        u16x8 blo[2], bhi[2];
        {
            const unsigned short* bp = Blds + (t0 * 16 + lm) * 40 + half * 8;
            blo[0] = *(const u16x8*)(bp);
            bhi[0] = *(const u16x8*)(bp + 16);
        }
#pragma unroll
        for (int t = 0; t < 7; ++t) {
            if (t < tcnt) {
                if (t + 1 < tcnt) {   // prefetch next fragment before the WMMA
                    const unsigned short* bp =
                        Blds + ((t0 + t + 1) * 16 + lm) * 40 + half * 8;
                    blo[(t + 1) & 1] = *(const u16x8*)(bp);
                    bhi[(t + 1) & 1] = *(const u16x8*)(bp + 16);
                }
                const bf16x16 bfrag = frag_cat(blo[t & 1], bhi[t & 1]);
                acc[t] = __builtin_amdgcn_wmma_f32_16x16x32_bf16(
                    false, afrag, false, bfrag, (short)0, acc[t], false, false);
            }
        }
        __syncthreads();               // all waves done reading this chunk

        // ---- scatter the register-staged next chunk into LDS ----
        if (hasNext) {
#pragma unroll
            for (int j = 0; j < BSTG; ++j) {
                const int gi  = tid + 256 * j;
                const int rem = (gi < WCHUNK) ? gi : gi - WCHUNK;
                const int kk  = rem / CH;
                const int n   = rem - kk * CH + ((gi < WCHUNK) ? 0 : CH);
                Blds[n * 40 + kk] = f2bf(bst[j]);
            }
#pragma unroll
            for (int j = 0; j < ASTG; ++j)
                Alds[(arow + 8 * j) * 40 + akk] = f2bf(ast[j]);
            __syncthreads();           // chunk visible before next compute
        }
    }

    // ---- epilogue: bias + store (C layout: VGPR r -> row half*8+r, col lane%16) ----
#pragma unroll
    for (int t = 0; t < 7; ++t) {
        if (t < tcnt) {
            const int n = (t0 + t) * 16 + lm;
            if (n < NCOLS) {
                const float bias = (n < CH) ? bbb[n] : bps[n - CH];
#pragma unroll
                for (int r = 0; r < 8; ++r) {
                    const int m = m0 + mt * 16 + half * 8 + r;
                    maps[(size_t)m * NCOLS + n] = acc[t][r] + bias;
                }
            }
        }
    }
}

// ---------------------------------------------------------------------------
// PS-ROI bilinear sample setup: p in [0,196) -> 4 tap bases + weights + bin
// ---------------------------------------------------------------------------
__device__ __forceinline__ void sample_setup(float rx1, float ry1, float rx2, float ry2,
                                             int p, int* base, float* w, int* cbin) {
    const int iy = p / HH;
    const int ix = p - iy * HH;
    const int by = iy >> 1, sy = iy & 1;
    const int bx = ix >> 1, sx = ix & 1;
    const float x1 = rx1 * 0.125f, y1 = ry1 * 0.125f;
    const float x2 = rx2 * 0.125f, y2 = ry2 * 0.125f;
    const float bw = (x2 - x1) * (1.0f / 7.0f);
    const float bh = (y2 - y1) * (1.0f / 7.0f);
    const float yy = y1 + ((float)by + ((float)sy + 0.5f) * 0.5f) * bh;
    const float xx = x1 + ((float)bx + ((float)sx + 0.5f) * 0.5f) * bw;
    const float y0f = floorf(yy), x0f = floorf(xx);
    const float wy = yy - y0f,    wx = xx - x0f;     // unclamped, as in reference
    int y0  = min(max((int)y0f, 0), HMAP - 1);
    int y1i = min(y0 + 1, HMAP - 1);
    int x0  = min(max((int)x0f, 0), WMAP - 1);
    int x1i = min(x0 + 1, WMAP - 1);
    base[0] = (y0  * WMAP + x0 ) * NCOLS;
    base[1] = (y0  * WMAP + x1i) * NCOLS;
    base[2] = (y1i * WMAP + x0 ) * NCOLS;
    base[3] = (y1i * WMAP + x1i) * NCOLS;
    w[0] = (1.0f - wy) * (1.0f - wx);
    w[1] = (1.0f - wy) * wx;
    w[2] = wy * (1.0f - wx);
    w[3] = wy * wx;
    *cbin = by * KK + bx;
}

__device__ __forceinline__ float bilin(const float* __restrict__ maps,
                                       const int* base, const float* w, int col) {
    return maps[base[0] + col] * w[0] + maps[base[1] + col] * w[1]
         + maps[base[2] + col] * w[2] + maps[base[3] + col] * w[3];
}

// ---------------------------------------------------------------------------
// Kernel B: per original ROI (300 blocks): pooled bbox deltas -> proposal,
// write rois_all rows [0..299]=rois, [300..599]=clipped proposals.
// ---------------------------------------------------------------------------
__global__ __launch_bounds__(256) void proposal_kernel(
    const float* __restrict__ maps, const float* __restrict__ rois,
    float* __restrict__ rois_all_ws, float* __restrict__ rois_all_out) {
    const int r   = blockIdx.x;
    const int tid = threadIdx.x;
    __shared__ float s_v[4 * NSAMP];

    const float rx1 = rois[r * 5 + 1], ry1 = rois[r * 5 + 2];
    const float rx2 = rois[r * 5 + 3], ry2 = rois[r * 5 + 4];

    if (tid < NSAMP) {
        int base[4]; float w[4]; int cbin;
        sample_setup(rx1, ry1, rx2, ry2, tid, base, w, &cbin);
#pragma unroll
        for (int g = 0; g < 4; ++g)
            s_v[g * NSAMP + tid] = bilin(maps, base, w, g * 49 + cbin);
    }
    __syncthreads();

    __shared__ float s_d[4];
    if (tid < 4) {                       // deterministic fixed-order reduction
        float s = 0.0f;
        for (int i = 0; i < NSAMP; ++i) s += s_v[tid * NSAMP + i];
        s_d[tid] = s * (1.0f / (float)NSAMP);
    }
    __syncthreads();

    if (tid == 0) {
        const float dx = s_d[0], dy = s_d[1], dw = s_d[2], dh = s_d[3];
        const float w0 = rx2 - rx1 + 1.0f, h0 = ry2 - ry1 + 1.0f;
        const float cx = rx1 + 0.5f * w0,  cy = ry1 + 0.5f * h0;
        const float pcx = dx * w0 + cx,    pcy = dy * h0 + cy;
        const float pw  = expf(dw) * w0,   ph  = expf(dh) * h0;
        float p1 = fminf(fmaxf(pcx - 0.5f * pw, 0.0f), IMM1);
        float p2 = fminf(fmaxf(pcy - 0.5f * ph, 0.0f), IMM1);
        float p3 = fminf(fmaxf(pcx + 0.5f * pw, 0.0f), IMM1);
        float p4 = fminf(fmaxf(pcy + 0.5f * ph, 0.0f), IMM1);
        float prop[5] = {0.0f, p1, p2, p3, p4};
#pragma unroll
        for (int c = 0; c < 5; ++c) {
            const float orig = rois[r * 5 + c];
            rois_all_ws [r * 5 + c] = orig;
            rois_all_out[r * 5 + c] = orig;
            rois_all_ws [(NROIS + r) * 5 + c] = prop[c];
            rois_all_out[(NROIS + r) * 5 + c] = prop[c];
        }
    }
}

// ---------------------------------------------------------------------------
// Kernel C: per rois_all row (600 blocks): bbox pooling + part2_1 heads.
// ---------------------------------------------------------------------------
__global__ __launch_bounds__(256) void heads_kernel(
    const float* __restrict__ maps, const float* __restrict__ rois_all,
    float* __restrict__ out, float* __restrict__ scores_ws) {
    const int r   = blockIdx.x;
    const int tid = threadIdx.x;
    __shared__ float s_bb[4 * NSAMP];
    __shared__ float s_cm[2 * NSAMP];
    __shared__ float s_red[6];
    __shared__ int   s_cls;

    const float rx1 = rois_all[r * 5 + 1], ry1 = rois_all[r * 5 + 2];
    const float rx2 = rois_all[r * 5 + 3], ry2 = rois_all[r * 5 + 4];

    float mk[2][2];   // mask[j][c] kept per thread for the final write
    mk[0][0] = mk[0][1] = mk[1][0] = mk[1][1] = 0.0f;

    if (tid < NSAMP) {
        int base[4]; float w[4]; int cbin;
        sample_setup(rx1, ry1, rx2, ry2, tid, base, w, &cbin);
        // bbox groups from bb_map (cols [0,196))
#pragma unroll
        for (int g = 0; g < 4; ++g)
            s_bb[g * NSAMP + tid] = bilin(maps, base, w, g * 49 + cbin);
        // ps groups from ps_map (cols [196,392))
        float mc[4];
#pragma unroll
        for (int g = 0; g < 4; ++g)
            mc[g] = bilin(maps, base, w, CH + g * 49 + cbin);
        // mask_cls[j][c] = mc[c*2 + j]; softmax over c; cls_max = max over c
#pragma unroll
        for (int j = 0; j < 2; ++j) {
            const float a0 = mc[0 * 2 + j], a1 = mc[1 * 2 + j];
            const float mx = fmaxf(a0, a1);
            const float e0 = expf(a0 - mx), e1 = expf(a1 - mx);
            const float inv = 1.0f / (e0 + e1);
            mk[j][0] = e0 * inv;
            mk[j][1] = e1 * inv;
            s_cm[j * NSAMP + tid] = mx;
        }
    }
    __syncthreads();

    if (tid < 4) {                        // bbox means (deterministic order)
        float s = 0.0f;
        for (int i = 0; i < NSAMP; ++i) s += s_bb[tid * NSAMP + i];
        s_red[tid] = s * (1.0f / (float)NSAMP);
    } else if (tid < 6) {                 // cls_ave[j]
        const int j = tid - 4;
        float s = 0.0f;
        for (int i = 0; i < NSAMP; ++i) s += s_cm[j * NSAMP + i];
        s_red[4 + j] = s * (1.0f / (float)NSAMP);
    }
    __syncthreads();

    if (tid == 0) {
#pragma unroll
        for (int g = 0; g < 4; ++g) out[OFF_BBOX + r * 4 + g] = s_red[g];
        const float a0 = s_red[4], a1 = s_red[5];
        const float mx = fmaxf(a0, a1);
        const float e0 = expf(a0 - mx), e1 = expf(a1 - mx);
        const float inv = 1.0f / (e0 + e1);
        const float c0 = e0 * inv, c1 = e1 * inv;
        const int   cr = (c1 > c0) ? 1 : 0;     // argmax, first index on ties
        const float cs = fmaxf(c0, c1);
        out[OFF_CLS   + r * 2 + 0] = c0;
        out[OFF_CLS   + r * 2 + 1] = c1;
        out[OFF_CLSR  + r] = (float)cr;
        out[OFF_SCORE + r] = cs;
        scores_ws[r] = cs;
        s_cls = cr;
    }
    __syncthreads();

    if (tid < NSAMP) {
        const int j = s_cls;
        const size_t o = OFF_MASK + ((size_t)r * NSAMP + tid) * 2;
        out[o + 0] = mk[j][0];
        out[o + 1] = mk[j][1];
    }
}

// ---------------------------------------------------------------------------
// Kernel D: NMS over 600 boxes (single block, matches reference semantics).
// ---------------------------------------------------------------------------
__global__ __launch_bounds__(256) void nms_kernel(
    const float* __restrict__ rois_all, const float* __restrict__ scores,
    float* __restrict__ keep_out) {
    __shared__ int   s_order[NALL];
    __shared__ float s_bx[NALL][4];
    __shared__ float s_area[NALL];
    __shared__ int   s_keep[NALL];
    __shared__ int   s_sup;
    const int tid = threadIdx.x;

    // stable descending argsort by rank counting
    for (int i = tid; i < NALL; i += 256) {
        const float si = scores[i];
        int rank = 0;
        for (int j = 0; j < NALL; ++j) {
            const float sj = scores[j];
            if (sj > si || (sj == si && j < i)) rank++;
        }
        s_order[rank] = i;
    }
    __syncthreads();

    for (int i = tid; i < NALL; i += 256) {
        const int o = s_order[i];
        const float b0 = rois_all[o * 5 + 1], b1 = rois_all[o * 5 + 2];
        const float b2 = rois_all[o * 5 + 3], b3 = rois_all[o * 5 + 4];
        s_bx[i][0] = b0; s_bx[i][1] = b1; s_bx[i][2] = b2; s_bx[i][3] = b3;
        s_area[i]  = fmaxf(b2 - b0, 0.0f) * fmaxf(b3 - b1, 0.0f);
        s_keep[i]  = 0;
    }
    __syncthreads();

    for (int i = 0; i < NALL; ++i) {
        if (tid == 0) s_sup = 0;
        __syncthreads();
        for (int j = tid; j < i; j += 256) {
            if (s_keep[j]) {
                const float ix1 = fmaxf(s_bx[i][0], s_bx[j][0]);
                const float iy1 = fmaxf(s_bx[i][1], s_bx[j][1]);
                const float ix2 = fminf(s_bx[i][2], s_bx[j][2]);
                const float iy2 = fminf(s_bx[i][3], s_bx[j][3]);
                const float inter = fmaxf(ix2 - ix1, 0.0f) * fmaxf(iy2 - iy1, 0.0f);
                const float iou = inter / (s_area[i] + s_area[j] - inter + 1e-8f);
                if (iou > IOU_T) atomicOr(&s_sup, 1);
            }
        }
        __syncthreads();
        if (tid == 0) s_keep[i] = (s_sup == 0) ? 1 : 0;
        __syncthreads();
    }

    for (int i = tid; i < MAXOUT; i += 256) keep_out[i] = -1.0f;
    __syncthreads();
    if (tid == 0) {
        int cnt = 0;
        for (int i = 0; i < NALL; ++i) {
            if (s_keep[i]) {
                if (cnt < MAXOUT) keep_out[cnt] = (float)s_order[i];
                cnt++;
            }
        }
    }
}

// ---------------------------------------------------------------------------
// Host launcher
// ---------------------------------------------------------------------------
extern "C" void kernel_launch(void* const* d_in, const int* in_sizes, int n_in,
                              void* d_out, int out_size, void* d_ws, size_t ws_size,
                              hipStream_t stream) {
    const float* feat = (const float*)d_in[0];   // [1,160,160,1024]
    const float* W_ps = (const float*)d_in[1];   // [1024,196]
    const float* b_ps = (const float*)d_in[2];   // [196]
    const float* W_bb = (const float*)d_in[3];   // [1024,196]
    const float* b_bb = (const float*)d_in[4];   // [196]
    const float* rois = (const float*)d_in[5];   // [300,5]
    float* out = (float*)d_out;

    float* maps        = (float*)d_ws;                       // [25600,392] = 40.1 MB
    float* rois_all_ws = maps + (size_t)MROWS * NCOLS;       // [600,5]
    float* scores_ws   = rois_all_ws + (size_t)NALL * 5;     // [600]

    gemm_maps_kernel<<<MROWS / 32, 256, 0, stream>>>(feat, W_bb, W_ps, b_bb, b_ps, maps);
    proposal_kernel<<<NROIS, 256, 0, stream>>>(maps, rois, rois_all_ws, out + OFF_ROIS);
    heads_kernel<<<NALL, 256, 0, stream>>>(maps, rois_all_ws, out, scores_ws);
    nms_kernel<<<1, 256, 0, stream>>>(rois_all_ws, scores_ws, out + OFF_KEEP);
}
